// GCNWithKE_64639257805298
// MI455X (gfx1250) — matrix operand order repaired and checked by reading, not compiled, not run on hardware
//
#include <hip/hip_runtime.h>
#include <hip/hip_bf16.h>

typedef _Float16 v16h __attribute__((ext_vector_type(16)));
typedef _Float16 v8h  __attribute__((ext_vector_type(8)));
typedef _Float16 h4   __attribute__((ext_vector_type(4)));
typedef float    v8f  __attribute__((ext_vector_type(8)));

#define IN_DIM 512
#define KE_DIM 2
#define HID    16
#define OUT    7
#define WPB    4   // waves per block in the WMMA GEMM

// ---------------------------------------------------------------------------
// Pre-swizzle W1[0:512,:] into f16 "B-fragment" order for v_wmma_f32_16x16x32_f16.
// For K-block blk (32 K-values), lane l, element e (0..15):
//   K = blk*32 + (l>=16 ? 16 : 0) + e,  N = l & 15
// blob[blk*512 + l*16 + e] = (f16) W1[K*16 + N]
// ---------------------------------------------------------------------------
__global__ __launch_bounds__(512) void prep_w1h(const float* __restrict__ W1,
                                                _Float16* __restrict__ W1h) {
    int t = threadIdx.x;            // 0..511 = 16 blocks x 32 lanes
    int blk  = t >> 5;
    int lane = t & 31;
    int kbase = blk * 32 + ((lane >> 4) << 4);
    int col   = lane & 15;
    _Float16* dst = W1h + (size_t)(blk * 32 + lane) * 16;
#pragma unroll
    for (int e = 0; e < 16; ++e)
        dst[e] = (_Float16)W1[(size_t)(kbase + e) * HID + col];
}

// ---------------------------------------------------------------------------
// In-degree (self loops folded in later as +1).
// ---------------------------------------------------------------------------
__global__ void deg_kernel(const int* __restrict__ dst, float* __restrict__ deg, int E) {
    int e = blockIdx.x * blockDim.x + threadIdx.x;
    if (e < E) atomicAdd(&deg[dst[e]], 1.0f);
}

__global__ void dinv_kernel(const float* __restrict__ deg, float* __restrict__ dinv, int N) {
    int i = blockIdx.x * blockDim.x + threadIdx.x;
    if (i < N) dinv[i] = rsqrtf(deg[i] + 1.0f);   // deg incl. self loop -> always > 0
}

// ---------------------------------------------------------------------------
// Layer-1 GEMM: h1[N,16] = [x | ke] @ W1  (no bias yet; bias after aggregation).
// One wave per 16-row tile. x tile staged fp32->f16 into LDS, 16 WMMA steps
// over K=512, then VALU fixup for the 2 ke columns on the f32 accumulator.
// ---------------------------------------------------------------------------
__global__ __launch_bounds__(WPB * 32) void gemm1_wmma(
    const float* __restrict__ x, const float* __restrict__ ke,
    const float* __restrict__ W1, const _Float16* __restrict__ W1h,
    float* __restrict__ h1, int N, int TB) {
    __shared__ h4 lds[WPB][16][IN_DIM / 4];   // 4 * 16 * 128 * 8B = 64 KB

    int wave = threadIdx.x >> 5;
    int lane = threadIdx.x & 31;
    int tile = blockIdx.x * WPB + wave;
    bool active = tile < TB;
    int rowbase = tile * 16;

    if (active) {
        // Cooperative, coalesced stage: 16 rows x 512 f32 -> f16 LDS tile.
        for (int i = 0; i < 64; ++i) {           // 2048 float4 / 32 lanes
            int idx = i * 32 + lane;
            int r   = idx >> 7;                  // row within tile
            int cv  = idx & 127;                 // float4 column
            int rr  = rowbase + r; if (rr >= N) rr = N - 1;
            float4 f = ((const float4*)(x + (size_t)rr * IN_DIM))[cv];
            h4 hv = { (_Float16)f.x, (_Float16)f.y, (_Float16)f.z, (_Float16)f.w };
            lds[wave][r][cv] = hv;
        }
    }
    __syncthreads();
    if (!active) return;

    int hi   = lane >> 4;        // lane group: 0 or 1
    int mrow = lane & 15;        // A-matrix row owned by this lane
    int col  = lane & 15;        // C/D column owned by this lane
    const _Float16* lrow = (const _Float16*)&lds[wave][mrow][0];

    v8f acc = {};
#pragma unroll
    for (int k0 = 0; k0 < IN_DIM; k0 += 32) {
        // A 16x32 f16, contiguous k order (same as W1h): lanes 0-15 elems = K {0..15}, lanes 16-31 = K {16..31}
        union { v16h v; struct { v8h l, h; } p; } a;
        int base = k0 + hi * 16;
        a.p.l = *(const v8h*)(lrow + base);
        a.p.h = *(const v8h*)(lrow + base + 8);
        v16h bfrag = *(const v16h*)(W1h + (size_t)(k0 >> 5) * 512 + lane * 16);
        acc = __builtin_amdgcn_wmma_f32_16x16x32_f16(
            /*neg_a=*/false, a.v, /*neg_b=*/false, bfrag,
            /*c_mod=*/(short)0, acc, /*reuse_a=*/false, /*reuse_b=*/false);
    }

    // K-remainder: the 2 ke columns (rows 512,513 of W1), exact fp32.
    float w512 = W1[(size_t)IN_DIM * HID + col];
    float w513 = W1[(size_t)(IN_DIM + 1) * HID + col];
#pragma unroll
    for (int v = 0; v < 8; ++v) {
        int row = rowbase + v + hi * 8;          // C/D layout: vgpr v, lane group hi
        if (row < N) {
            float2 k2 = *(const float2*)(ke + (size_t)row * KE_DIM);
            h1[(size_t)row * HID + col] = acc[v] + k2.x * w512 + k2.y * w513;
        }
    }
}

// ---------------------------------------------------------------------------
// Edge scatter, 16 channels: agg1[v] += dinv[u]*dinv[v]*h1[u]
// ---------------------------------------------------------------------------
__global__ void agg16_kernel(const int* __restrict__ src, const int* __restrict__ dst,
                             const float* __restrict__ dinv, const float* __restrict__ h1,
                             float* __restrict__ agg, int E) {
    int e = blockIdx.x * blockDim.x + threadIdx.x;
    if (e >= E) return;
    int u = src[e], v = dst[e];
    float w = dinv[u] * dinv[v];
    const float4* hu = (const float4*)(h1 + (size_t)u * HID);
    float* av = agg + (size_t)v * HID;
#pragma unroll
    for (int q = 0; q < 4; ++q) {
        float4 f = hu[q];
        atomicAdd(av + q * 4 + 0, w * f.x);
        atomicAdd(av + q * 4 + 1, w * f.y);
        atomicAdd(av + q * 4 + 2, w * f.z);
        atomicAdd(av + q * 4 + 3, w * f.w);
    }
}

// Self loop + bias + ReLU (in place on agg1): h = relu(agg1 + dinv^2*h1 + b1)
__global__ void self_bias_relu(float* __restrict__ agg1, const float* __restrict__ h1,
                               const float* __restrict__ dinv, const float* __restrict__ b1,
                               int total) {
    int t = blockIdx.x * blockDim.x + threadIdx.x;
    if (t >= total) return;
    int i = t >> 4, c = t & 15;
    float d2 = dinv[i] * dinv[i];
    float val = agg1[t] + d2 * h1[t] + b1[c];
    agg1[t] = val > 0.0f ? val : 0.0f;
}

// ---------------------------------------------------------------------------
// Layer-2 GEMM: h2[N,7] = h @ W2 (16x7, staged in LDS; no bias yet).
// ---------------------------------------------------------------------------
__global__ __launch_bounds__(256) void gemm2_kernel(const float* __restrict__ h,
                                                    const float* __restrict__ W2,
                                                    float* __restrict__ h2, int N) {
    __shared__ float w2s[HID * OUT];
    if (threadIdx.x < HID * OUT) w2s[threadIdx.x] = W2[threadIdx.x];
    __syncthreads();
    int i = blockIdx.x * blockDim.x + threadIdx.x;
    if (i >= N) return;
    float hr[HID];
    const float4* hp = (const float4*)(h + (size_t)i * HID);
#pragma unroll
    for (int q = 0; q < 4; ++q) {
        float4 f = hp[q];
        hr[q * 4 + 0] = f.x; hr[q * 4 + 1] = f.y; hr[q * 4 + 2] = f.z; hr[q * 4 + 3] = f.w;
    }
#pragma unroll
    for (int o = 0; o < OUT; ++o) {
        float s = 0.0f;
#pragma unroll
        for (int k = 0; k < HID; ++k) s = fmaf(hr[k], w2s[k * OUT + o], s);
        h2[(size_t)i * OUT + o] = s;
    }
}

// Edge scatter, 7 channels
__global__ void agg7_kernel(const int* __restrict__ src, const int* __restrict__ dst,
                            const float* __restrict__ dinv, const float* __restrict__ h2,
                            float* __restrict__ agg, int E) {
    int e = blockIdx.x * blockDim.x + threadIdx.x;
    if (e >= E) return;
    int u = src[e], v = dst[e];
    float w = dinv[u] * dinv[v];
    const float* hu = h2 + (size_t)u * OUT;
    float* av = agg + (size_t)v * OUT;
#pragma unroll
    for (int c = 0; c < OUT; ++c) atomicAdd(av + c, w * hu[c]);
}

// Self loop + bias + log_softmax -> out
__global__ void final_kernel(const float* __restrict__ agg2, const float* __restrict__ h2,
                             const float* __restrict__ dinv, const float* __restrict__ b2,
                             float* __restrict__ out, int N) {
    int i = blockIdx.x * blockDim.x + threadIdx.x;
    if (i >= N) return;
    float d2 = dinv[i] * dinv[i];
    float z[OUT];
#pragma unroll
    for (int c = 0; c < OUT; ++c)
        z[c] = agg2[(size_t)i * OUT + c] + d2 * h2[(size_t)i * OUT + c] + b2[c];
    float m = z[0];
#pragma unroll
    for (int c = 1; c < OUT; ++c) m = fmaxf(m, z[c]);
    float s = 0.0f;
#pragma unroll
    for (int c = 0; c < OUT; ++c) s += expf(z[c] - m);
    float lse = logf(s);
#pragma unroll
    for (int c = 0; c < OUT; ++c) out[(size_t)i * OUT + c] = z[c] - m - lse;
}

// ---------------------------------------------------------------------------
extern "C" void kernel_launch(void* const* d_in, const int* in_sizes, int n_in,
                              void* d_out, int out_size, void* d_ws, size_t ws_size,
                              hipStream_t stream) {
    const float* x  = (const float*)d_in[0];
    const float* ke = (const float*)d_in[1];
    const int*   ei = (const int*)d_in[2];
    const float* W1 = (const float*)d_in[3];
    const float* b1 = (const float*)d_in[4];
    const float* W2 = (const float*)d_in[5];
    const float* b2 = (const float*)d_in[6];
    (void)n_in; (void)out_size; (void)ws_size; (void)b1;

    int N = in_sizes[0] / IN_DIM;
    int E = in_sizes[2] / 2;
    const int* srcI = ei;        // edge_index[0]
    const int* dstI = ei + E;    // edge_index[1]
    float* out = (float*)d_out;

    // Workspace carve-up (~19.2 MB for N=100k)
    char* ws = (char*)d_ws;
    float* deg      = (float*)ws;  ws += (size_t)N * 4;
    float* dinv     = (float*)ws;  ws += (size_t)N * 4;
    _Float16* W1h   = (_Float16*)ws; ws += (size_t)16 * 512 * 2;     // 16 KB blob
    float* h1       = (float*)ws;  ws += (size_t)N * HID * 4;
    float* agg1     = (float*)ws;  ws += (size_t)N * HID * 4;        // becomes h after relu
    float* h2       = (float*)ws;  ws += (size_t)N * OUT * 4;
    float* agg2     = (float*)ws;  ws += (size_t)N * OUT * 4;

    hipMemsetAsync(deg,  0, (size_t)N * 4,        stream);
    hipMemsetAsync(agg1, 0, (size_t)N * HID * 4,  stream);
    hipMemsetAsync(agg2, 0, (size_t)N * OUT * 4,  stream);

    prep_w1h<<<1, 512, 0, stream>>>(W1, W1h);
    deg_kernel<<<(E + 255) / 256, 256, 0, stream>>>(dstI, deg, E);
    dinv_kernel<<<(N + 255) / 256, 256, 0, stream>>>(deg, dinv, N);

    int TB = (N + 15) / 16;
    gemm1_wmma<<<(TB + WPB - 1) / WPB, WPB * 32, 0, stream>>>(x, ke, W1, W1h, h1, N, TB);

    agg16_kernel<<<(E + 255) / 256, 256, 0, stream>>>(srcI, dstI, dinv, h1, agg1, E);
    int tot16 = N * HID;
    self_bias_relu<<<(tot16 + 255) / 256, 256, 0, stream>>>(agg1, h1, dinv, b1, tot16);

    gemm2_kernel<<<(N + 255) / 256, 256, 0, stream>>>(agg1, W2, h2, N);
    agg7_kernel<<<(E + 255) / 256, 256, 0, stream>>>(srcI, dstI, dinv, h2, agg2, E);
    final_kernel<<<(N + 255) / 256, 256, 0, stream>>>(agg2, h2, dinv, b2, out, N);
}
